// CharRNN_652835029793
// MI455X (gfx1250) — matrix operand order repaired and checked by reading, not compiled, not run on hardware
//
#include <hip/hip_runtime.h>

typedef __attribute__((ext_vector_type(16))) __bf16 v16bf;
typedef __attribute__((ext_vector_type(8)))  __bf16 v8bf;
typedef __attribute__((ext_vector_type(8)))  float  v8f;

#define T_SEQ 2048
#define BB    64
#define XX    256
#define HH    1024
#define G4    4096          // 4*H
#define KTOT  1280          // X + H
#define NWG   32
#define NTHR  256

// ---------------- workspace layout (bytes) ----------------
// Wg   : swizzled gates B  [40 kt][256 nt][32 lane][16 e] bf16 = 10485760
// Wout : swizzled head  B  [32 kt][ 16 nt][32 lane][16 e] bf16 =   524288
// bg   : permuted gate bias fp32[4096]                         =    16384
// hb0/1: ping-pong h state bf16 [64][1024]                     =   131072 each
// cnt  : grid barrier counter
// xbf  : optional bf16 copy of input [T][B][X]                 = 67108864
#define WS_WG    0
#define WS_WOUT  10485760
#define WS_BG    (WS_WOUT + 524288)
#define WS_HB0   (WS_BG + 16384)
#define WS_HB1   (WS_HB0 + 131072)
#define WS_CNT   (WS_HB1 + 131072)
#define WS_XBF   (WS_CNT + 1024)
#define XBF_BYTES ((size_t)T_SEQ*BB*XX*2)

// ---------------- prep: convert + swizzle weights (runs every launch) ----
__global__ void __launch_bounds__(NTHR)
prep_kernel(const float* __restrict__ W_ih, const float* __restrict__ W_hh,
            const float* __restrict__ b_ih, const float* __restrict__ b_hh,
            const float* __restrict__ W_out, const float* __restrict__ h0,
            __bf16* __restrict__ Wg, __bf16* __restrict__ Wout,
            float* __restrict__ bg, __bf16* __restrict__ hb0, int* __restrict__ cnt)
{
  long idx = (long)blockIdx.x * NTHR + threadIdx.x;
  const long NWGT = (long)KTOT * G4;      // 5242880
  const long NWO  = (long)HH * XX;        // 262144
  if (idx < NWGT) {
    // swizzled B layout: value = B[k][n], k = kt*32 + (lane>=16)*8 + (e&7) + (e>=8)*16, n = nt*16 + (lane&15)
    int e    = (int)( idx        & 15);
    int lane = (int)((idx >> 4)  & 31);
    int nt   = (int)((idx >> 9)  & 255);
    int kt   = (int)( idx >> 17);
    int k = kt*32 + ((lane>>4)<<3) + (e & 7) + ((e >> 3) << 4);
    int n = nt*16 + (lane & 15);
    // column permutation: n = blk*128 + g*32 + ul  ->  gate g of hidden unit blk*32+ul
    int blk = n >> 7, rem = n & 127, g = rem >> 5, ul = rem & 31;
    int srow = g*HH + blk*32 + ul;        // PyTorch gate order i,f,g,o
    float v = (k < XX) ? W_ih[srow*XX + k] : W_hh[(size_t)srow*HH + (k - XX)];
    Wg[idx] = (__bf16)v;
  } else if (idx < NWGT + NWO) {
    long j = idx - NWGT;
    int e    = (int)( j        & 15);
    int lane = (int)((j >> 4)  & 31);
    int nt   = (int)((j >> 9)  & 15);
    int kt   = (int)( j >> 13);
    int k = kt*32 + ((lane>>4)<<3) + (e & 7) + ((e >> 3) << 4);
    int n = nt*16 + (lane & 15);
    Wout[j] = (__bf16)W_out[(size_t)n*HH + k];   // B[k][n] = W_out[n][k]
  } else if (idx < NWGT + NWO + G4) {
    int p = (int)(idx - NWGT - NWO);
    int blk = p >> 7, rem = p & 127, g = rem >> 5, ul = rem & 31;
    int srow = g*HH + blk*32 + ul;
    bg[p] = b_ih[srow] + b_hh[srow];
  } else if (idx < NWGT + NWO + G4 + (long)BB*HH) {
    int p = (int)(idx - NWGT - NWO - G4);
    hb0[p] = (__bf16)h0[p];
  } else if (idx == NWGT + NWO + G4 + (long)BB*HH) {
    *cnt = 0;
  }
}

// optional: one-shot bf16 copy of the whole input sequence
__global__ void __launch_bounds__(NTHR)
xconv_kernel(const float* __restrict__ x, __bf16* __restrict__ xbf)
{
  long idx = (long)blockIdx.x * NTHR + threadIdx.x;
  if (idx < (long)T_SEQ*BB*XX) xbf[idx] = (__bf16)x[idx];
}

// ---------------- A-tile loaders (documented 16-bit A layout) ------------
// lane 0-15: row M=lane, K = kb..kb+7 and kb+16..kb+23 ; lane 16-31: kb += 8
__device__ __forceinline__ v16bf load_a_bf(const __bf16* __restrict__ base,
                                           int row, int stride, int kb) {
  const __bf16* p = base + (size_t)row*stride + kb;
  v8bf lo = *(const v8bf*)(p);
  v8bf hi = *(const v8bf*)(p + 16);
  v16bf a;
  #pragma unroll
  for (int i = 0; i < 8; ++i) { a[i] = lo[i]; a[i+8] = hi[i]; }
  return a;
}
__device__ __forceinline__ v16bf load_a_f32(const float* __restrict__ base,
                                            int row, int stride, int kb) {
  const float* p = base + (size_t)row*stride + kb;
  v8f lo = *(const v8f*)(p);
  v8f hi = *(const v8f*)(p + 16);
  v16bf a;
  #pragma unroll
  for (int i = 0; i < 8; ++i) { a[i] = (__bf16)lo[i]; a[i+8] = (__bf16)hi[i]; }
  return a;
}
__device__ __forceinline__ float sigmoidf_(float x) { return 1.0f/(1.0f+__expf(-x)); }

// ---------------- persistent LSTM kernel, 32 WGs ------------------------
__global__ void __launch_bounds__(NTHR)
lstm_kernel(const float* __restrict__ xin, const __bf16* __restrict__ xbf,
            const int use_xbf,
            const float* __restrict__ c0,
            const float* __restrict__ bg, const float* __restrict__ b_out,
            const __bf16* __restrict__ Wg, const __bf16* __restrict__ Wout,
            __bf16* __restrict__ hb0, __bf16* __restrict__ hb1,
            int* __restrict__ cnt, float* __restrict__ out)
{
  __shared__ float gsm[64][128];   // gates slice (i|f|g|o x 32 units)
  __shared__ float cls[64][32];    // persistent cell state slice
  __shared__ float ysm[16][256];   // head logits (rotating WG quad)
  __shared__ float red[16][16];
  __shared__ float rmax[16];
  __shared__ float rsum[16];

  const int tid   = threadIdx.x;
  const int lane  = tid & 31;
  const int w     = tid >> 5;          // wave id 0..7
  const int wg    = blockIdx.x;        // 0..31
  const int laneN = lane & 15;
  const int kbhi  = (lane >> 4) << 3;  // 0 or 8 (A K-offset; also C M-offset)
  const int u0    = wg * 32;           // hidden-unit slice base

  float* outY = out;
  float* outH = out + (size_t)T_SEQ*BB*XX;
  float* outC = outH + (size_t)BB*HH;

  const float bias = bg[wg*128 + w*16 + laneN];
  const float bo0  = b_out[(w*2    )*16 + laneN];
  const float bo1  = b_out[(w*2 + 1)*16 + laneN];

  for (int i = tid; i < 64*32; i += NTHR)
    cls[i>>5][i&31] = c0[(size_t)(i>>5)*HH + u0 + (i&31)];
  __syncthreads();

  for (int t = 0; t < T_SEQ; ++t) {
    const __bf16* hr = (t & 1) ? hb1 : hb0;
    __bf16*       hw = (t & 1) ? hb0 : hb1;

    // ===== Phase A: gates[64x128] = [x_t | h] @ Wg(:, wg-slice) =====
    v8f acc[4] = {};
    if (use_xbf) {                                 // input part, pre-converted bf16
      const __bf16* xtb = xbf + (size_t)t*BB*XX;
      for (int kt = 0; kt < 8; ++kt) {
        const __bf16* bp = Wg + ((size_t)(kt*256 + wg*8 + w)*32 + lane)*16;
        v16bf b = *(const v16bf*)bp;
        __builtin_prefetch(bp + 256*32*16, 0, 1);
        #pragma unroll
        for (int m = 0; m < 4; ++m) {
          v16bf a = load_a_bf(xtb, m*16 + laneN, XX, kt*32 + kbhi);
          acc[m] = __builtin_amdgcn_wmma_f32_16x16x32_bf16(
                     false, a, false, b, (short)0, acc[m], false, false);
        }
      }
    } else {                                       // fallback: convert on the fly
      const float* xt = xin + (size_t)t*BB*XX;
      for (int kt = 0; kt < 8; ++kt) {
        const __bf16* bp = Wg + ((size_t)(kt*256 + wg*8 + w)*32 + lane)*16;
        v16bf b = *(const v16bf*)bp;
        __builtin_prefetch(bp + 256*32*16, 0, 1);
        #pragma unroll
        for (int m = 0; m < 4; ++m) {
          v16bf a = load_a_f32(xt, m*16 + laneN, XX, kt*32 + kbhi);
          acc[m] = __builtin_amdgcn_wmma_f32_16x16x32_bf16(
                     false, a, false, b, (short)0, acc[m], false, false);
        }
      }
    }
    for (int kt = 8; kt < 40; ++kt) {              // hidden part, K=256..1279
      const __bf16* bp = Wg + ((size_t)(kt*256 + wg*8 + w)*32 + lane)*16;
      v16bf b = *(const v16bf*)bp;
      __builtin_prefetch(bp + 256*32*16, 0, 1);
      #pragma unroll
      for (int m = 0; m < 4; ++m) {
        v16bf a = load_a_bf(hr, m*16 + laneN, HH, (kt-8)*32 + kbhi);
        acc[m] = __builtin_amdgcn_wmma_f32_16x16x32_bf16(
                   false, a, false, b, (short)0, acc[m], false, false);
      }
    }
    // C layout: VGPR r -> M = r (lanes 0-15) / 8+r (lanes 16-31); N = lane&15
    #pragma unroll
    for (int m = 0; m < 4; ++m)
      #pragma unroll
      for (int r = 0; r < 8; ++r)
        gsm[m*16 + r + kbhi][w*16 + laneN] = acc[m][r] + bias;
    __syncthreads();

    // ===== LSTM cell update for this WG's 32 hidden units =====
    const bool last = (t == T_SEQ - 1);
    for (int i = tid; i < 64*32; i += NTHR) {
      int row = i >> 5, u = i & 31;
      float ig = sigmoidf_(gsm[row][      u]);
      float fg = sigmoidf_(gsm[row][ 32 + u]);
      float gg = tanhf    (gsm[row][ 64 + u]);
      float og = sigmoidf_(gsm[row][ 96 + u]);
      float cn = fg * cls[row][u] + ig * gg;
      float hn = og * tanhf(cn);
      cls[row][u] = cn;
      hw[(size_t)row*HH + u0 + u] = (__bf16)hn;
      if (last) {
        outH[(size_t)row*HH + u0 + u] = hn;
        outC[(size_t)row*HH + u0 + u] = cn;
      }
    }

    // ===== grid barrier (one per step; h is double-buffered) =====
    __syncthreads();
    if (tid == 0) {
      __threadfence();
      atomicAdd(cnt, 1);
      volatile int* vc = cnt;
      const int target = (t + 1) * NWG;
      while (*vc < target) __builtin_amdgcn_s_sleep(8);
    }
    __syncthreads();
    __threadfence();

    // ===== Phase B: y = softmax(h_new @ Wout + b_out) =====
    // round-robin: quad (t&7) of WGs handles it this step, 16 batch rows each
    if ((wg >> 2) == (t & 7)) {
      const int m0 = (wg & 3) * 16;
      v8f a2[2] = {};
      for (int kt = 0; kt < 32; ++kt) {
        v16bf a = load_a_bf(hw, m0 + laneN, HH, kt*32 + kbhi);
        #pragma unroll
        for (int j = 0; j < 2; ++j) {
          const __bf16* bp = Wout + ((size_t)(kt*16 + w*2 + j)*32 + lane)*16;
          v16bf b = *(const v16bf*)bp;
          a2[j] = __builtin_amdgcn_wmma_f32_16x16x32_bf16(
                    false, a, false, b, (short)0, a2[j], false, false);
        }
      }
      #pragma unroll
      for (int j = 0; j < 2; ++j) {
        float bo = j ? bo1 : bo0;
        #pragma unroll
        for (int r = 0; r < 8; ++r)
          ysm[r + kbhi][(w*2 + j)*16 + laneN] = a2[j][r] + bo;
      }
      __syncthreads();
      // row softmax: 16 threads per row, 16 cols each
      int row = tid >> 4, seg = tid & 15;
      float mx = -3.4e38f;
      #pragma unroll
      for (int c = 0; c < 16; ++c) mx = fmaxf(mx, ysm[row][seg*16 + c]);
      red[row][seg] = mx;
      __syncthreads();
      if (seg == 0) {
        float m2 = red[row][0];
        #pragma unroll
        for (int i2 = 1; i2 < 16; ++i2) m2 = fmaxf(m2, red[row][i2]);
        rmax[row] = m2;
      }
      __syncthreads();
      float s = 0.f;
      #pragma unroll
      for (int c = 0; c < 16; ++c) s += __expf(ysm[row][seg*16 + c] - rmax[row]);
      red[row][seg] = s;
      __syncthreads();
      if (seg == 0) {
        float s2 = 0.f;
        #pragma unroll
        for (int i2 = 0; i2 < 16; ++i2) s2 += red[row][i2];
        rsum[row] = s2;
      }
      __syncthreads();
      float inv = 1.0f / rsum[row];
      float* yp = outY + ((size_t)t*BB + m0 + row)*XX + seg*16;
      #pragma unroll
      for (int c = 0; c < 16; ++c)
        yp[c] = __expf(ysm[row][seg*16 + c] - rmax[row]) * inv;
      __syncthreads();
    }
  }
}

extern "C" void kernel_launch(void* const* d_in, const int* in_sizes, int n_in,
                              void* d_out, int out_size, void* d_ws, size_t ws_size,
                              hipStream_t stream) {
  (void)in_sizes; (void)n_in; (void)out_size;
  const float* x     = (const float*)d_in[0];
  const float* h0    = (const float*)d_in[1];
  const float* c0    = (const float*)d_in[2];
  const float* W_ih  = (const float*)d_in[3];
  const float* W_hh  = (const float*)d_in[4];
  const float* b_ih  = (const float*)d_in[5];
  const float* b_hh  = (const float*)d_in[6];
  const float* W_out = (const float*)d_in[7];
  const float* b_out = (const float*)d_in[8];

  char* ws = (char*)d_ws;
  __bf16* Wg   = (__bf16*)(ws + WS_WG);
  __bf16* Wout = (__bf16*)(ws + WS_WOUT);
  float*  bg   = (float*) (ws + WS_BG);
  __bf16* hb0  = (__bf16*)(ws + WS_HB0);
  __bf16* hb1  = (__bf16*)(ws + WS_HB1);
  int*    cnt  = (int*)   (ws + WS_CNT);
  __bf16* xbf  = (__bf16*)(ws + WS_XBF);

  // launch-time constant: is there room for the bf16 input copy?
  const int use_xbf = (ws_size >= (size_t)WS_XBF + XBF_BYTES) ? 1 : 0;

  const long total = (long)KTOT*G4 + (long)HH*XX + G4 + (long)BB*HH + 1;
  const int prep_blocks = (int)((total + NTHR - 1) / NTHR);
  prep_kernel<<<prep_blocks, NTHR, 0, stream>>>(W_ih, W_hh, b_ih, b_hh, W_out, h0,
                                                Wg, Wout, bg, hb0, cnt);
  if (use_xbf) {
    const long nx = (long)T_SEQ*BB*XX;
    xconv_kernel<<<(int)((nx + NTHR - 1) / NTHR), NTHR, 0, stream>>>(x, xbf);
  }
  lstm_kernel<<<NWG, NTHR, 0, stream>>>(x, xbf, use_xbf, c0, bg, b_out, Wg, Wout,
                                        hb0, hb1, cnt, (float*)d_out);
}
